// TextRNN_63848983823091
// MI455X (gfx1250) — matrix operand order repaired and checked by reading, not compiled, not run on hardware
//
#include <hip/hip_runtime.h>
#include <hip/hip_bf16.h>

// ---------------------------------------------------------------------------
// Bi-LSTM text classifier for MI455X (gfx1250, wave32, WMMA).
// B=64 T=512 V=50000 E=128 H=256 G=4H=1024 O=14
// ---------------------------------------------------------------------------

#define BB 64
#define TT 512
#define EE 128
#define HH 256
#define GG 1024
#define OO 14

typedef __attribute__((ext_vector_type(4)))  unsigned int u32x4;
typedef __attribute__((ext_vector_type(16))) __bf16       v16bf;
typedef __attribute__((ext_vector_type(8)))  float        v8f;

union FragAB { u32x4 q[2]; v16bf v; };          // 16 bf16 = one WMMA A/B fragment per lane
union Pack8  { u32x4 q;    unsigned short s[8]; };

__device__ __forceinline__ unsigned short f2bf_us(float f) {
  unsigned u = __builtin_bit_cast(unsigned, f);
  return (unsigned short)((u + 0x7FFFu + ((u >> 16) & 1u)) >> 16);  // RNE
}
__device__ __forceinline__ float bf2f_us(unsigned short s) {
  return __builtin_bit_cast(float, ((unsigned)s) << 16);
}
__device__ __forceinline__ float sigf(float x) {
  return 1.0f / (1.0f + __expf(-x));
}
__device__ __forceinline__ float tanhfast(float x) {
  float e = __expf(2.0f * x);
  return 1.0f - 2.0f / (e + 1.0f);
}

// ---------------------------------------------------------------------------
// Kernel 1: convert weights to bf16, combine biases (bih+bhh) per direction.
// ---------------------------------------------------------------------------
__global__ void prep_kernel(const float* wih_f, const float* whh_f,
                            const float* bih_f, const float* bhh_f,
                            const float* wih_b, const float* whh_b,
                            const float* bih_b, const float* bhh_b,
                            unsigned short* wihbf, unsigned short* whhbf,
                            float* biasc) {
  const int NW = GG * EE;     // 131072
  const int NH = GG * HH;     // 262144
  int idx = blockIdx.x * 256 + threadIdx.x;
  if (idx < 2 * NW) {
    int d = idx / NW, j = idx - d * NW;
    wihbf[idx] = f2bf_us(d ? wih_b[j] : wih_f[j]);
  } else if (idx < 2 * NW + 2 * NH) {
    int k = idx - 2 * NW;
    int d = k / NH, j = k - d * NH;
    whhbf[k] = f2bf_us(d ? whh_b[j] : whh_f[j]);
  } else if (idx < 2 * NW + 2 * NH + 2 * GG) {
    int k = idx - 2 * NW - 2 * NH;
    int d = k / GG, j = k - d * GG;
    biasc[k] = d ? (bih_b[j] + bhh_b[j]) : (bih_f[j] + bhh_f[j]);
  }
}

// ---------------------------------------------------------------------------
// Kernel 2: embedding gather + input-gate GEMM (both directions).
//   xg[d][t][n][b] = sum_e emb[tok[b,t],e] * wih[d][n,e] + bih[d][n] + bhh[d][n]
// One block = one (t, b-tile of 16) x 8 col-tiles (1 col-tile per wave).
// grid: (2048 row-tiles, 8 col-groups, 2 dirs), block 256 (8 wave32).
// ---------------------------------------------------------------------------
__global__ void __launch_bounds__(256)
xgemm_kernel(const int* __restrict__ tokens, const float* __restrict__ emb,
             const unsigned short* __restrict__ wihbf,
             const float* __restrict__ biasc,
             unsigned short* __restrict__ xg) {
  const int q  = blockIdx.x;          // 0..2047
  const int cg = blockIdx.y;          // 0..7
  const int d  = blockIdx.z;          // 0..1
  const int t  = q >> 2;
  const int b0 = (q & 3) << 4;

  __shared__ int            tok[16];
  __shared__ unsigned short xt[16 * EE];   // 16 x 128 bf16 A-tile

  const int tid = threadIdx.x;
  if (tid < 16) tok[tid] = tokens[(b0 + tid) * TT + t];
  __syncthreads();
  {
    int row = tid >> 4;
    int c0  = (tid & 15) << 3;
    const float* er = emb + (size_t)tok[row] * EE + c0;
#pragma unroll
    for (int j = 0; j < 8; ++j) xt[row * EE + c0 + j] = f2bf_us(er[j]);
  }
  __syncthreads();

  const int w    = tid >> 5;
  const int lane = tid & 31;
  const int nl   = lane & 15;
  const int half = lane >> 4;
  const int off  = half << 3;    // C/D row offset (M = r + off)
  const int kbA  = half << 3;    // A K-base within chunk
  const int kbB  = half << 3;    // B K-base within chunk
  const int n_g  = (cg * 8 + w) * 16 + nl;

  float bv = biasc[d * GG + n_g];
  v8f acc;
#pragma unroll
  for (int r = 0; r < 8; ++r) acc[r] = bv;

  const unsigned short* wbase = wihbf + (size_t)d * GG * EE + (size_t)n_g * EE;
#pragma unroll
  for (int kk = 0; kk < 4; ++kk) {            // K = 128 = 4 x 32
    FragAB a, b;
    const unsigned short* xr = xt + nl * EE + 32 * kk + kbA;
    a.q[0] = *(const u32x4*)xr;
    a.q[1] = *(const u32x4*)(xr + 16);
    const unsigned short* wr = wbase + 32 * kk + kbB;
    b.q[0] = *(const u32x4*)wr;
    b.q[1] = *(const u32x4*)(wr + 16);
    acc = __builtin_amdgcn_wmma_f32_16x16x32_bf16(false, a.v, false, b.v,
                                                  (short)0, acc, false, false);
  }

  Pack8 p;
#pragma unroll
  for (int r = 0; r < 8; ++r) p.s[r] = f2bf_us(acc[r]);
  size_t oidx = (((size_t)(d * TT + t)) * GG + n_g) * BB + b0 + off;
  *(u32x4*)(xg + oidx) = p.q;
}

// ---------------------------------------------------------------------------
// Kernel 3: sequential LSTM recurrence. 8 blocks = 4 batch-tiles x 2 dirs.
// Per step: gates = xg[t] + h @ Whh.T via bf16 WMMA (K=256), then gate math;
// h lives in LDS (16x256 bf16), c stays in registers for all 512 steps.
// Wave w owns gate-col tiles {w+8s}: matching i/f/g/o columns -> local update.
// ---------------------------------------------------------------------------
__global__ void __launch_bounds__(256)
recur_kernel(const int* __restrict__ lengths,
             const unsigned short* __restrict__ whhbf,
             const unsigned short* __restrict__ xg,
             float* __restrict__ last) {
  const int blk = blockIdx.x;       // 0..7
  const int d   = blk >> 2;
  const int b0  = (blk & 3) << 4;

  __shared__ unsigned short hbuf[16 * HH];   // 8 KB: h state, bf16

  const int tid = threadIdx.x;
#pragma unroll
  for (int j = 0; j < 16; ++j) hbuf[tid + j * 256] = 0;  // h0 = 0

  const int w    = tid >> 5;
  const int lane = tid & 31;
  const int nl   = lane & 15;
  const int half = lane >> 4;
  const int off  = half << 3;
  const int kbA  = half << 3;
  const int kbB  = half << 3;

  int lenr[8];
#pragma unroll
  for (int r = 0; r < 8; ++r) lenr[r] = lengths[b0 + off + r];

  v8f acc[8];
  v8f cc[2] = {};                       // cell state: 2 groups x 8 rows, f32
  const unsigned short* whbase = whhbf + (size_t)d * GG * HH;

  __syncthreads();

  for (int step = 0; step < TT; ++step) {
    const int t = d ? (TT - 1 - step) : step;

    // --- init accumulators from precomputed xg (includes biases) ---
#pragma unroll
    for (int s = 0; s < 8; ++s) {
      int n_g = (w + 8 * s) * 16 + nl;
      const unsigned short* xp =
          xg + (((size_t)(d * TT + t)) * GG + n_g) * BB + b0 + off;
      Pack8 hx;
      hx.q = *(const u32x4*)xp;
#pragma unroll
      for (int r = 0; r < 8; ++r) acc[s][r] = bf2f_us(hx.s[r]);
    }
    // prefetch next step's xg into cache hierarchy
    if (step + 1 < TT) {
      int tn = d ? (TT - 2 - step) : (step + 1);
      __builtin_prefetch(
          xg + (((size_t)(d * TT + tn)) * GG + w * 16 + nl) * BB + b0 + off,
          0, 0);
    }

    // --- gates += h @ Whh.T : K=256 in 8 WMMA k-chunks ---
#pragma unroll
    for (int kk = 0; kk < 8; ++kk) {
      FragAB a;
      const unsigned short* hr = hbuf + nl * HH + 32 * kk + kbA;
      a.q[0] = *(const u32x4*)hr;
      a.q[1] = *(const u32x4*)(hr + 16);
#pragma unroll
      for (int s = 0; s < 8; ++s) {
        int n_g = (w + 8 * s) * 16 + nl;
        FragAB b;
        const unsigned short* wr = whbase + (size_t)n_g * HH + 32 * kk + kbB;
        b.q[0] = *(const u32x4*)wr;
        b.q[1] = *(const u32x4*)(wr + 16);
        acc[s] = __builtin_amdgcn_wmma_f32_16x16x32_bf16(
            false, a.v, false, b.v, (short)0, acc[s], false, false);
      }
    }
    __syncthreads();   // all waves done reading old h

    // --- gate nonlinearity + state update (acc order: i,i,f,f,g,g,o,o) ---
#pragma unroll
    for (int X = 0; X < 2; ++X) {
      const int hc = X * 128 + 16 * w + nl;   // h column owned by this lane
#pragma unroll
      for (int r = 0; r < 8; ++r) {
        float iv = sigf(acc[0 + X][r]);
        float fv = sigf(acc[2 + X][r]);
        float gv = tanhfast(acc[4 + X][r]);
        float ov = sigf(acc[6 + X][r]);
        float cn = fv * cc[X][r] + iv * gv;
        cc[X][r] = cn;
        float hv = ov * tanhfast(cn);
        hbuf[(r + off) * HH + hc] = f2bf_us(hv);
        if (t == lenr[r] - 1)
          last[(size_t)(b0 + off + r) * (2 * HH) + d * HH + hc] = hv;
      }
    }
    __syncthreads();   // new h visible before next step's A-fragment loads
  }
}

// ---------------------------------------------------------------------------
// Kernel 4: final FC [64,512]x[512,14] + log_softmax. One block, 896 threads.
// ---------------------------------------------------------------------------
__global__ void head_kernel(const float* __restrict__ last,
                            const float* __restrict__ fcw,
                            const float* __restrict__ fcb,
                            float* __restrict__ out) {
  __shared__ float lg[BB * OO];
  __shared__ float stat[BB];
  const int tid = threadIdx.x;
  if (tid < BB * OO) {
    int b = tid / OO, o = tid - b * OO;
    const float* lp = last + (size_t)b * (2 * HH);
    const float* wp = fcw + (size_t)o * (2 * HH);
    float s = fcb[o];
#pragma unroll 8
    for (int k = 0; k < 2 * HH; ++k) s += lp[k] * wp[k];
    lg[tid] = s;
  }
  __syncthreads();
  if (tid < BB) {
    float m = -1e30f;
#pragma unroll
    for (int o = 0; o < OO; ++o) m = fmaxf(m, lg[tid * OO + o]);
    float sum = 0.0f;
#pragma unroll
    for (int o = 0; o < OO; ++o) sum += __expf(lg[tid * OO + o] - m);
    stat[tid] = m + __logf(sum);
  }
  __syncthreads();
  if (tid < BB * OO) {
    int b = tid / OO;
    out[tid] = lg[tid] - stat[b];
  }
}

// ---------------------------------------------------------------------------
// Launch
// ---------------------------------------------------------------------------
extern "C" void kernel_launch(void* const* d_in, const int* in_sizes, int n_in,
                              void* d_out, int out_size, void* d_ws,
                              size_t ws_size, hipStream_t stream) {
  (void)in_sizes; (void)n_in; (void)out_size; (void)ws_size;
  const int*   tokens  = (const int*)d_in[0];
  const int*   lengths = (const int*)d_in[1];
  const float* emb     = (const float*)d_in[2];
  const float* wih_f   = (const float*)d_in[3];
  const float* whh_f   = (const float*)d_in[4];
  const float* bih_f   = (const float*)d_in[5];
  const float* bhh_f   = (const float*)d_in[6];
  const float* wih_b   = (const float*)d_in[7];
  const float* whh_b   = (const float*)d_in[8];
  const float* bih_b   = (const float*)d_in[9];
  const float* bhh_b   = (const float*)d_in[10];
  const float* fc_w    = (const float*)d_in[11];
  const float* fc_b    = (const float*)d_in[12];

  // Workspace layout (bytes)
  const size_t XG_BYTES   = (size_t)2 * TT * GG * BB * 2;   // 134,217,728
  const size_t WHH_BYTES  = (size_t)2 * GG * HH * 2;        //   1,048,576
  const size_t WIH_BYTES  = (size_t)2 * GG * EE * 2;        //     524,288
  const size_t BIAS_BYTES = (size_t)2 * GG * 4;             //       8,192

  char* ws = (char*)d_ws;
  unsigned short* xg     = (unsigned short*)(ws);
  unsigned short* whhbf  = (unsigned short*)(ws + XG_BYTES);
  unsigned short* wihbf  = (unsigned short*)(ws + XG_BYTES + WHH_BYTES);
  float*          biasc  = (float*)(ws + XG_BYTES + WHH_BYTES + WIH_BYTES);
  float*          lastb  = (float*)(ws + XG_BYTES + WHH_BYTES + WIH_BYTES + BIAS_BYTES);

  // 1) weight conversion: 2*131072 + 2*262144 + 2048 = 788480 items
  prep_kernel<<<788480 / 256, 256, 0, stream>>>(
      wih_f, whh_f, bih_f, bhh_f, wih_b, whh_b, bih_b, bhh_b,
      wihbf, whhbf, biasc);

  // 2) embedding + input-gate GEMM -> xg (bf16, [dir][T][4H][B])
  xgemm_kernel<<<dim3(2048, 8, 2), 256, 0, stream>>>(
      tokens, emb, wihbf, biasc, xg);

  // 3) sequential bi-LSTM recurrence (8 WGPs: 4 batch-tiles x 2 dirs)
  recur_kernel<<<8, 256, 0, stream>>>(lengths, whhbf, xg, lastb);

  // 4) classifier head + log_softmax
  head_kernel<<<1, 896, 0, stream>>>(lastb, fc_w, fc_b, (float*)d_out);
}
